// GroupedQueryAttention_64252710748465
// MI455X (gfx1250) — matrix operand order repaired and checked by reading, not compiled, not run on hardware
//
#include <hip/hip_runtime.h>

typedef __attribute__((ext_vector_type(16))) _Float16 v16h;
typedef __attribute__((ext_vector_type(8)))  _Float16 v8h;
typedef __attribute__((ext_vector_type(4)))  _Float16 v4h;
typedef __attribute__((ext_vector_type(8)))  float    v8f;
typedef __attribute__((ext_vector_type(4)))  float    v4f;

#define WMMA_F16(a, b, c) \
  __builtin_amdgcn_wmma_f32_16x16x32_f16(false, (a), false, (b), (short)0, (c), false, false)

constexpr int S_LEN    = 2048;
constexpr int D_MODEL  = 2048;
constexpr int NUM_HEADS = 32;
constexpr int NUM_KV    = 8;
constexpr int D_K       = 64;
constexpr int KV_DIM    = NUM_KV * D_K;   // 512

// ---------------------------------------------------------------------------
// Fragment loaders per CDNA5 WMMA VGPR layouts (CDNA5 ISA §7.12.2)
// A (16x32 f16): lane L -> row = L&15 ; halves a[0..7]=K[base..base+7],
//                a[8..15]=K[base+16..base+23], base = (L>=16) ? 8 : 0
// B (32x16 f16) from Bt (N x K row-major): lane L -> col = L&15 ;
//                b[0..7]=K[base..base+7], b[8..15]=K[base+16..base+23], base = (L>=16) ? 8 : 0
// ---------------------------------------------------------------------------
__device__ __forceinline__ v16h load_a16(const _Float16* __restrict__ row, int kbase) {
  v8h lo = *(const v8h*)(row + kbase);
  v8h hi = *(const v8h*)(row + kbase + 16);
  return __builtin_shufflevector(lo, hi, 0,1,2,3,4,5,6,7,8,9,10,11,12,13,14,15);
}

__device__ __forceinline__ v16h load_a32(const float* __restrict__ row, int kbase) {
  v4f f0 = *(const v4f*)(row + kbase);
  v4f f1 = *(const v4f*)(row + kbase + 4);
  v4f f2 = *(const v4f*)(row + kbase + 16);
  v4f f3 = *(const v4f*)(row + kbase + 20);
  v16h t;
#pragma unroll
  for (int e = 0; e < 4; ++e) {
    t[e]      = (_Float16)f0[e];
    t[e + 4]  = (_Float16)f1[e];
    t[e + 8]  = (_Float16)f2[e];
    t[e + 12] = (_Float16)f3[e];
  }
  return t;
}

__device__ __forceinline__ v16h load_b16(const _Float16* __restrict__ brow, int kbase) {
  v8h lo = *(const v8h*)(brow + kbase);
  v8h hi = *(const v8h*)(brow + kbase + 16);
  return __builtin_shufflevector(lo, hi, 0,1,2,3,4,5,6,7,8,9,10,11,12,13,14,15);
}

// ---------------------------------------------------------------------------
// Generic WMMA GEMM:  C(M x N, f32, row-major) = A(M x K) * Bt(N x K)^T
// Wave tile 64x64 (4x4 16x16x32 WMMA frags); block = WM x WN waves.
// AF32: A is fp32, converted to f16 in the fragment loader (used for attn@V).
// ---------------------------------------------------------------------------
template <int WM, int WN, bool AF32>
__global__ __launch_bounds__(32 * WM * WN)
void gemm_wmma_k(const void* __restrict__ Av, int lda, long long strideA,
                 const _Float16* __restrict__ Btb, int ldb, long long strideB, int bdiv,
                 float* __restrict__ Cb, int ldc, long long strideC, int K) {
  const int lane = threadIdx.x & 31;
  const int wave = threadIdx.x >> 5;
  const int wm = wave / WN, wn = wave % WN;
  const int bm = blockIdx.x * (WM * 64) + wm * 64;
  const int bn = blockIdx.y * (WN * 64) + wn * 64;
  const int bz = blockIdx.z;

  const float*    A32 = (const float*)Av    + (long long)bz * strideA;
  const _Float16* A16 = (const _Float16*)Av + (long long)bz * strideA;
  const _Float16* Bt  = Btb + (long long)(bz / bdiv) * strideB;
  float* C = Cb + (long long)bz * strideC;

  const int rowA = bm + (lane & 15);
  const int aoff = (lane >> 4) << 3;   // 0 or 8
  const int colB = bn + (lane & 15);
  const int boff = (lane >> 4) << 3;   // 0 or 8

  v8f acc[4][4];
#pragma unroll
  for (int i = 0; i < 4; ++i)
#pragma unroll
    for (int j = 0; j < 4; ++j)
#pragma unroll
      for (int e = 0; e < 8; ++e) acc[i][j][e] = 0.0f;

  for (int k0 = 0; k0 < K; k0 += 32) {
    if (k0 + 32 < K) {  // prefetch next K tile (global_prefetch_b8)
      if (AF32) __builtin_prefetch(A32 + (long long)rowA * lda + k0 + 32 + aoff, 0, 3);
      else      __builtin_prefetch(A16 + (long long)rowA * lda + k0 + 32 + aoff, 0, 3);
      __builtin_prefetch(Bt + (long long)colB * ldb + k0 + 32 + boff, 0, 3);
    }
    v16h a[4], b[4];
#pragma unroll
    for (int mi = 0; mi < 4; ++mi) {
      const long long r = (long long)(rowA + mi * 16);
      if (AF32) a[mi] = load_a32(A32 + r * lda, k0 + aoff);
      else      a[mi] = load_a16(A16 + r * lda, k0 + aoff);
    }
#pragma unroll
    for (int ni = 0; ni < 4; ++ni)
      b[ni] = load_b16(Bt + (long long)(colB + ni * 16) * ldb, k0 + boff);
#pragma unroll
    for (int mi = 0; mi < 4; ++mi)
#pragma unroll
      for (int ni = 0; ni < 4; ++ni)
        acc[mi][ni] = WMMA_F16(a[mi], b[ni], acc[mi][ni]);
  }

  const int g8 = (lane >> 4) << 3;
#pragma unroll
  for (int mi = 0; mi < 4; ++mi)
#pragma unroll
    for (int ni = 0; ni < 4; ++ni)
#pragma unroll
      for (int r = 0; r < 8; ++r) {
        const int row = bm + mi * 16 + g8 + r;
        const int col = bn + ni * 16 + (lane & 15);
        C[(long long)row * ldc + col] = acc[mi][ni][r];
      }
}

// ---------------------------------------------------------------------------
// Fused scores + softmax: one block owns 32 full rows of one head.
// 16 waves x (32 rows x 128 cols) register-resident score slices -> WMMA.
// All 8 B-fragments of a k-step are preloaded into b[8] (one load clause,
// drained with partial s_wait_loadcnt while WMMAs retire) -- acc(128) +
// b(64) + a(16) ~ 220 VGPRs, still inside the 256-VGPR window.
// Row max/sum via lane shuffles + LDS cross-wave reduction; normalized attn
// written to global exactly once.
// ---------------------------------------------------------------------------
__global__ __launch_bounds__(512)
void attn_scores_softmax(const _Float16* __restrict__ Qh,   // [H][S][64]
                         const _Float16* __restrict__ Kh,   // [Hkv][S][64]
                         const float* __restrict__ mask,    // [S][S]
                         float* __restrict__ attn) {        // [H][S][S]
  const int lane = threadIdx.x & 31;
  const int w    = threadIdx.x >> 5;   // 0..15 : 128-wide column slice
  const int rb   = blockIdx.x;         // 32-row block
  const int h    = blockIdx.y;
  const int kvh  = h >> 2;             // GROUPS = 4

  const _Float16* Q  = Qh + (long long)h   * S_LEN * D_K;
  const _Float16* Kk = Kh + (long long)kvh * S_LEN * D_K;
  float* P = attn + (long long)h * S_LEN * S_LEN;

  const int g8   = (lane >> 4) << 3;
  const int rowA = rb * 32 + (lane & 15);
  const int colB = w * 128 + (lane & 15);
  const int boff = (lane >> 4) << 3;

  v8f acc[2][8];
#pragma unroll
  for (int mi = 0; mi < 2; ++mi)
#pragma unroll
    for (int ni = 0; ni < 8; ++ni)
#pragma unroll
      for (int e = 0; e < 8; ++e) acc[mi][ni][e] = 0.0f;

#pragma unroll
  for (int ks = 0; ks < 2; ++ks) {     // K = d_k = 64 -> two k-steps
    const int k0 = ks * 32;
    v16h a0 = load_a16(Q + (long long)rowA * D_K, k0 + g8);
    v16h a1 = load_a16(Q + (long long)(rowA + 16) * D_K, k0 + g8);
    v16h b[8];
#pragma unroll
    for (int ni = 0; ni < 8; ++ni)
      b[ni] = load_b16(Kk + (long long)(colB + ni * 16) * D_K, k0 + boff);
#pragma unroll
    for (int ni = 0; ni < 8; ++ni) {
      acc[0][ni] = WMMA_F16(a0, b[ni], acc[0][ni]);
      acc[1][ni] = WMMA_F16(a1, b[ni], acc[1][ni]);
    }
  }

  // scale + mask, per-lane partial row max
  float pmax[2][8];
#pragma unroll
  for (int mi = 0; mi < 2; ++mi)
#pragma unroll
    for (int r = 0; r < 8; ++r) pmax[mi][r] = -3.0e38f;

#pragma unroll
  for (int mi = 0; mi < 2; ++mi)
#pragma unroll
    for (int ni = 0; ni < 8; ++ni)
#pragma unroll
      for (int r = 0; r < 8; ++r) {
        const int row = rb * 32 + mi * 16 + g8 + r;
        const int col = w * 128 + ni * 16 + (lane & 15);
        float v = acc[mi][ni][r] * 0.125f +
                  mask[(long long)row * S_LEN + col] * (-1e9f);
        acc[mi][ni][r] = v;
        pmax[mi][r] = fmaxf(pmax[mi][r], v);
      }

  __shared__ float red[16][32];
  __shared__ float rowred[32];

  // --- row max: shuffle-reduce across 16 lanes sharing a row, then LDS across waves
#pragma unroll
  for (int mi = 0; mi < 2; ++mi)
#pragma unroll
    for (int r = 0; r < 8; ++r) {
      float v = pmax[mi][r];
#pragma unroll
      for (int off = 1; off < 16; off <<= 1) v = fmaxf(v, __shfl_xor(v, off, 32));
      pmax[mi][r] = v;
    }
  if ((lane & 15) == 0) {
#pragma unroll
    for (int mi = 0; mi < 2; ++mi)
#pragma unroll
      for (int r = 0; r < 8; ++r) red[w][mi * 16 + g8 + r] = pmax[mi][r];
  }
  __syncthreads();
  if (threadIdx.x < 32) {
    float m = -3.0e38f;
#pragma unroll
    for (int ww = 0; ww < 16; ++ww) m = fmaxf(m, red[ww][threadIdx.x]);
    rowred[threadIdx.x] = m;
  }
  __syncthreads();
  float rmax[2][8];
#pragma unroll
  for (int mi = 0; mi < 2; ++mi)
#pragma unroll
    for (int r = 0; r < 8; ++r) rmax[mi][r] = rowred[mi * 16 + g8 + r];

  // --- exp + row sum
  float psum[2][8];
#pragma unroll
  for (int mi = 0; mi < 2; ++mi)
#pragma unroll
    for (int r = 0; r < 8; ++r) psum[mi][r] = 0.0f;
#pragma unroll
  for (int mi = 0; mi < 2; ++mi)
#pragma unroll
    for (int ni = 0; ni < 8; ++ni)
#pragma unroll
      for (int r = 0; r < 8; ++r) {
        const float e = __expf(acc[mi][ni][r] - rmax[mi][r]);
        acc[mi][ni][r] = e;
        psum[mi][r] += e;
      }
#pragma unroll
  for (int mi = 0; mi < 2; ++mi)
#pragma unroll
    for (int r = 0; r < 8; ++r) {
      float v = psum[mi][r];
#pragma unroll
      for (int off = 1; off < 16; off <<= 1) v += __shfl_xor(v, off, 32);
      psum[mi][r] = v;
    }
  if ((lane & 15) == 0) {
#pragma unroll
    for (int mi = 0; mi < 2; ++mi)
#pragma unroll
      for (int r = 0; r < 8; ++r) red[w][mi * 16 + g8 + r] = psum[mi][r];
  }
  __syncthreads();
  if (threadIdx.x < 32) {
    float s = 0.0f;
#pragma unroll
    for (int ww = 0; ww < 16; ++ww) s += red[ww][threadIdx.x];
    rowred[threadIdx.x] = s;
  }
  __syncthreads();
  float rinv[2][8];
#pragma unroll
  for (int mi = 0; mi < 2; ++mi)
#pragma unroll
    for (int r = 0; r < 8; ++r) rinv[mi][r] = 1.0f / rowred[mi * 16 + g8 + r];

  // --- normalized attention, written once
#pragma unroll
  for (int mi = 0; mi < 2; ++mi)
#pragma unroll
    for (int ni = 0; ni < 8; ++ni)
#pragma unroll
      for (int r = 0; r < 8; ++r) {
        const int row = rb * 32 + mi * 16 + g8 + r;
        const int col = w * 128 + ni * 16 + (lane & 15);
        P[(long long)row * S_LEN + col] = acc[mi][ni][r] * rinv[mi][r];
      }
}

// ---------------------------------------------------------------------------
// Elementwise fp32 -> f16 (x4 vectorized)
// ---------------------------------------------------------------------------
__global__ __launch_bounds__(256)
void cvt_f32_f16(const float* __restrict__ in, _Float16* __restrict__ out, int n) {
  const int i = (blockIdx.x * 256 + threadIdx.x) * 4;
  if (i < n) {
    v4f x = *(const v4f*)(in + i);
    v4h y;
#pragma unroll
    for (int e = 0; e < 4; ++e) y[e] = (_Float16)x[e];
    *(v4h*)(out + i) = y;
  }
}

// Transpose + convert: Wt[n][k] = (f16) W[k][n]   (LDS-tiled, 32x32)
__global__ __launch_bounds__(256)
void transpose_cvt(const float* __restrict__ W, _Float16* __restrict__ Wt, int K, int N) {
  __shared__ float tile[32][33];
  const int kb = blockIdx.x * 32, nb = blockIdx.y * 32;
  for (int i = threadIdx.y; i < 32; i += 8)
    tile[i][threadIdx.x] = W[(long long)(kb + i) * N + nb + threadIdx.x];
  __syncthreads();
  for (int i = threadIdx.y; i < 32; i += 8)
    Wt[(long long)(nb + i) * K + kb + threadIdx.x] = (_Float16)tile[threadIdx.x][i];
}

// RoPE + head split: out[h][s][d] = (f16)(x*cos + rotate_half(x)*sin)
__global__ __launch_bounds__(64)
void rope_reshape(const float* __restrict__ X, int ld, _Float16* __restrict__ outHSd) {
  __shared__ float xs[64];
  const int d = threadIdx.x, s = blockIdx.x, h = blockIdx.y;
  const float x = X[(long long)s * ld + h * D_K + d];
  xs[d] = x;
  __syncthreads();
  const float partner = (d < 32) ? xs[d + 32] : xs[d - 32];
  const float rot = (d < 32) ? -partner : partner;
  // theta_i = 10000^(-2i/64) = exp(-i * ln(10000)/32), i = d>>1 (precise expf:
  // theta error is amplified by pos up to 2048 in the angle)
  const float theta = expf(-(float)(d >> 1) * 0.28782313662425574f);
  const float ang = (float)s * theta;
  float sn, c;
  sincosf(ang, &sn, &c);  // precise range reduction (ang up to ~2048 rad)
  outHSd[((long long)h * S_LEN + s) * D_K + d] = (_Float16)(x * c + rot * sn);
}

// V transpose per KV head: Vt[h][d][s] = (f16) V[s][h*64+d]
__global__ __launch_bounds__(256)
void vtrans(const float* __restrict__ V, _Float16* __restrict__ Vt) {
  __shared__ float tile[32][33];
  const int h = blockIdx.z;
  const int sb = blockIdx.x * 32, db = blockIdx.y * 32;
  for (int i = threadIdx.y; i < 32; i += 8)
    tile[i][threadIdx.x] = V[(long long)(sb + i) * KV_DIM + h * D_K + db + threadIdx.x];
  __syncthreads();
  for (int i = threadIdx.y; i < 32; i += 8)
    Vt[((long long)h * D_K + db + i) * S_LEN + sb + threadIdx.x] =
        (_Float16)tile[threadIdx.x][i];
}

// ---------------------------------------------------------------------------
extern "C" void kernel_launch(void* const* d_in, const int* in_sizes, int n_in,
                              void* d_out, int out_size, void* d_ws, size_t ws_size,
                              hipStream_t stream) {
  (void)in_sizes; (void)n_in; (void)out_size; (void)ws_size;

  const float* q    = (const float*)d_in[0];
  const float* k    = (const float*)d_in[1];
  const float* v    = (const float*)d_in[2];
  const float* mask = (const float*)d_in[3];
  const float* wq   = (const float*)d_in[4];
  const float* wk   = (const float*)d_in[5];
  const float* wv   = (const float*)d_in[6];
  const float* wo   = (const float*)d_in[7];

  float* out  = (float*)d_out;                          // (S, D_MODEL)
  float* attn = out + (size_t)S_LEN * D_MODEL;          // (H, S, S)

  char* ws = (char*)d_ws;
  size_t off = 0;
  auto take = [&](size_t bytes) -> char* {
    char* p = ws + off;
    off = (off + bytes + 255) & ~(size_t)255;
    return p;
  };

  _Float16* Xq16 = (_Float16*)take((size_t)S_LEN * D_MODEL * 2);
  _Float16* Xk16 = (_Float16*)take((size_t)S_LEN * D_MODEL * 2);
  _Float16* Xv16 = (_Float16*)take((size_t)S_LEN * D_MODEL * 2);
  _Float16* WqT  = (_Float16*)take((size_t)D_MODEL * D_MODEL * 2);
  _Float16* WkT  = (_Float16*)take((size_t)KV_DIM * D_MODEL * 2);
  _Float16* WvT  = (_Float16*)take((size_t)KV_DIM * D_MODEL * 2);
  _Float16* WoT  = (_Float16*)take((size_t)D_MODEL * D_MODEL * 2);
  float*    Q32  = (float*)take((size_t)S_LEN * D_MODEL * 4);
  float*    K32  = (float*)take((size_t)S_LEN * KV_DIM * 4);
  float*    V32  = (float*)take((size_t)S_LEN * KV_DIM * 4);
  _Float16* Q16  = (_Float16*)take((size_t)S_LEN * D_MODEL * 2);  // [H][S][64]
  _Float16* K16  = (_Float16*)take((size_t)S_LEN * KV_DIM * 2);   // [Hkv][S][64]
  _Float16* V16t = (_Float16*)take((size_t)S_LEN * KV_DIM * 2);   // [Hkv][64][S]
  float*    ctx32 = (float*)take((size_t)S_LEN * D_MODEL * 4);    // (S, D_MODEL)
  _Float16* ctx16 = (_Float16*)take((size_t)S_LEN * D_MODEL * 2);

  const int n1 = S_LEN * D_MODEL;  // 4M elements
  cvt_f32_f16<<<n1 / 1024, 256, 0, stream>>>(q, Xq16, n1);
  cvt_f32_f16<<<n1 / 1024, 256, 0, stream>>>(k, Xk16, n1);
  cvt_f32_f16<<<n1 / 1024, 256, 0, stream>>>(v, Xv16, n1);

  dim3 tb(32, 8);
  transpose_cvt<<<dim3(D_MODEL / 32, D_MODEL / 32), tb, 0, stream>>>(wq, WqT, D_MODEL, D_MODEL);
  transpose_cvt<<<dim3(D_MODEL / 32, KV_DIM / 32),  tb, 0, stream>>>(wk, WkT, D_MODEL, KV_DIM);
  transpose_cvt<<<dim3(D_MODEL / 32, KV_DIM / 32),  tb, 0, stream>>>(wv, WvT, D_MODEL, KV_DIM);
  transpose_cvt<<<dim3(D_MODEL / 32, D_MODEL / 32), tb, 0, stream>>>(wo, WoT, D_MODEL, D_MODEL);

  // Projections (f16 WMMA, f32 accumulate)
  gemm_wmma_k<2, 2, false><<<dim3(S_LEN / 128, D_MODEL / 128, 1), 128, 0, stream>>>(
      Xq16, D_MODEL, 0, WqT, D_MODEL, 0, 1, Q32, D_MODEL, 0, D_MODEL);
  gemm_wmma_k<2, 2, false><<<dim3(S_LEN / 128, KV_DIM / 128, 1), 128, 0, stream>>>(
      Xk16, D_MODEL, 0, WkT, D_MODEL, 0, 1, K32, KV_DIM, 0, D_MODEL);
  gemm_wmma_k<2, 2, false><<<dim3(S_LEN / 128, KV_DIM / 128, 1), 128, 0, stream>>>(
      Xv16, D_MODEL, 0, WvT, D_MODEL, 0, 1, V32, KV_DIM, 0, D_MODEL);

  rope_reshape<<<dim3(S_LEN, NUM_HEADS), 64, 0, stream>>>(Q32, D_MODEL, Q16);
  rope_reshape<<<dim3(S_LEN, NUM_KV),    64, 0, stream>>>(K32, KV_DIM,  K16);
  vtrans<<<dim3(S_LEN / 32, D_K / 32, NUM_KV), tb, 0, stream>>>(V32, V16t);

  // Fused scores (+scale, +mask) + softmax, single attn write
  attn_scores_softmax<<<dim3(S_LEN / 32, NUM_HEADS), 512, 0, stream>>>(Q16, K16, mask, attn);

  // ctx = attn @ V  (fp32 A converted to f16 in fragment loader; N=64 -> 4x1 waves)
  gemm_wmma_k<4, 1, true><<<dim3(S_LEN / 256, 1, NUM_HEADS), 128, 0, stream>>>(
      attn, S_LEN, (long long)S_LEN * S_LEN,
      V16t, S_LEN, (long long)D_K * S_LEN, 4,
      ctx32, D_MODEL, D_K, S_LEN);

  cvt_f32_f16<<<n1 / 1024, 256, 0, stream>>>(ctx32, ctx16, n1);

  // out = ctx @ wo
  gemm_wmma_k<2, 2, false><<<dim3(S_LEN / 128, D_MODEL / 128, 1), 128, 0, stream>>>(
      ctx16, D_MODEL, 0, WoT, D_MODEL, 0, 1, out, D_MODEL, 0, D_MODEL);
}
